// STConvBlock_24335284699207
// MI455X (gfx1250) — matrix operand order repaired and checked by reading, not compiled, not run on hardware
//
#include <hip/hip_runtime.h>
#include <hip/hip_bf16.h>
#include <stdint.h>

// ---------------------------------------------------------------------------
// STConvBlock for MI455X (gfx1250, wave32, WMMA + TDM).
// Dominant compute: Chebyshev spatial matmul, restructured as
//   out_s(b,t) = Abig(1024 x 4096) @ Hbig_bt(4096 x 64) + cheb_b
// with Abig = [T_0 | T_1 | T_2 | T_3] (bf16, L2-resident, 8 MB) and
// Hbig_bt rows (k*1024+m) = h(b,t,m,:) @ cheb_w[k]  (bf16).
// GEMM runs on v_wmma_f32_16x16x32_bf16; B tiles are DMA'd into LDS by the
// Tensor Data Mover (tensor_load_to_lds, TENSORcnt), double buffered.
// ---------------------------------------------------------------------------

typedef __bf16 bf16_t;
typedef __bf16 v16bf __attribute__((ext_vector_type(16)));
typedef float  v8f   __attribute__((ext_vector_type(8)));
typedef uint32_t u32x4 __attribute__((ext_vector_type(4)));
typedef uint32_t u32x8 __attribute__((ext_vector_type(8)));

#define B_    16
#define T_    64
#define N_    1024
#define FIN   32
#define FS    64
#define FO    64
#define KORD  4               // K_ORDER + 1
#define BT    (B_ * T_)       // 1024
#define KBIG  (KORD * N_)     // 4096 (reduction dim of the big GEMM)

// ---------------------------------------------------------------------------
// Kernel 0: cheb_basis (4,1024,1024) f32  ->  Abig (1024 x 4096) bf16
// Abig[n][k*1024+m] = basis[k][n][m]   (row-major: A fragments load contiguous)
// ---------------------------------------------------------------------------
__global__ __launch_bounds__(256) void k_convert_basis(
    const float* __restrict__ basis, bf16_t* __restrict__ Abig) {
  int o = blockIdx.x * 256 + threadIdx.x;          // 0 .. 4M-1
  int n  = o >> 12;                                 // row
  int kk = o & 4095;                                // col = k*1024 + m
  int k  = kk >> 10;
  int m  = kk & 1023;
  Abig[(size_t)n * KBIG + kk] = (bf16_t)basis[(size_t)k * N_ * N_ + (size_t)n * N_ + m];
}

// ---------------------------------------------------------------------------
// Kernel 1: temporal conv1 (causal, k=3) + GLU, fused with h @ cheb_w[k] fold.
// One thread per (bt, m) node. Writes HbigT column-major:
//   HbigT[bt][i][k*1024+m]   (i = output feature 0..63)
// so WMMA B-fragments (two runs of 8 consecutive K per lane) are contiguous bf16.
// ---------------------------------------------------------------------------
__global__ __launch_bounds__(256) void k_temporal1_fold(
    const float* __restrict__ x,        // [BT][N][FIN]
    const float* __restrict__ conv1_w,  // [128][FIN][3]
    const float* __restrict__ conv1_b,  // [128]
    const float* __restrict__ cheb_w,   // [KORD][FS][FS]
    bf16_t* __restrict__ HbigT) {       // [BT][FS][KBIG]
  extern __shared__ float smem[];
  float* sW  = smem;                     // 128*32*3 = 12288
  float* sB  = sW + 12288;               // 128
  float* sCW = sB + 128;                 // 4*64*64 = 16384
  for (int i = threadIdx.x; i < 12288; i += 256) sW[i] = conv1_w[i];
  for (int i = threadIdx.x; i < 128;   i += 256) sB[i] = conv1_b[i];
  for (int i = threadIdx.x; i < 16384; i += 256) sCW[i] = cheb_w[i];
  __syncthreads();

  int gid = blockIdx.x * 256 + threadIdx.x;  // 0 .. BT*N-1
  int bt = gid >> 10;
  int m  = gid & 1023;
  int t  = bt & (T_ - 1);

  // load 3 taps of x (zero left-pad)
  float xv[3][FIN];
#pragma unroll
  for (int j = 0; j < 3; ++j) {
    int tt = t - 2 + j;
    if (tt >= 0) {
      const float4* xp = (const float4*)(x + ((size_t)(bt - (2 - j)) * N_ + m) * FIN);
#pragma unroll
      for (int q = 0; q < FIN / 4; ++q) {
        float4 v = xp[q];
        xv[j][q * 4 + 0] = v.x; xv[j][q * 4 + 1] = v.y;
        xv[j][q * 4 + 2] = v.z; xv[j][q * 4 + 3] = v.w;
      }
    } else {
#pragma unroll
      for (int f = 0; f < FIN; ++f) xv[j][f] = 0.f;
    }
  }

  // conv + GLU:  h[i] = a[i] * sigmoid(g[i]),  a = ch 0..63, g = ch 64..127
  float h[FS];
#pragma unroll 1
  for (int i = 0; i < FS; ++i) {
    float a = sB[i], g = sB[FS + i];
    const float* wa = sW + i * (FIN * 3);
    const float* wg = sW + (FS + i) * (FIN * 3);
#pragma unroll
    for (int f = 0; f < FIN; ++f)
#pragma unroll
      for (int j = 0; j < 3; ++j) {
        float xf = xv[j][f];
        a = fmaf(wa[f * 3 + j], xf, a);
        g = fmaf(wg[f * 3 + j], xf, g);
      }
    h[i] = a * (1.0f / (1.0f + __expf(-g)));
  }

  // fold: HbigT[bt][i][k*1024+m] = sum_j h[j] * cheb_w[k][j][i]
  bf16_t* outp = HbigT + (size_t)bt * FS * KBIG + m;
#pragma unroll 1
  for (int k = 0; k < KORD; ++k) {
    const float* wk = sCW + k * FS * FS;
#pragma unroll 1
    for (int i = 0; i < FS; ++i) {
      float s = 0.f;
#pragma unroll 8
      for (int j = 0; j < FS; ++j) s = fmaf(h[j], wk[j * FS + i], s);
      outp[(size_t)i * KBIG + k * N_] = (bf16_t)s;
    }
  }
}

// ---------------------------------------------------------------------------
// Kernel 2: the big GEMM on WMMA with TDM-staged B.
// Grid: (BT, N/128). Block: 256 threads = 8 wave32.
// Each wave computes a 16x64 tile (4 accumulators of 16x16 f32).
// B (Hbig_bt) is DMA'd into LDS by the Tensor Data Mover in K-chunks of 128,
// double buffered, with D# pad fields inserting 16B per 256B row so the 16
// column-lanes of a fragment hit distinct bank groups (row pitch 272B).
// A fragments (shared Abig, L2-resident) loaded straight from global:
//   lane layout (16-bit A 16x32): lane<16 -> row l, K {0..7, 16..23};
//   lane>=16 -> row l-16, K {8..15, 24..31}  => two contiguous b128 loads.
// B lane layout: lane<16 -> col l, K {0..7, 16..23}; lane>=16 -> col l-16, K {8..15, 24..31}
//   => two contiguous b128 LDS reads (HbigT is column-major in K).
// ---------------------------------------------------------------------------
#define CKB  128
#define CKP  (CKB + 8)

// Issue one TDM 2D tile load: tile 128(K, contiguous) x 64(cols), element 2B,
// tensor_dim0 = 4096 (stride between columns in HbigT), LDS pad: 4 DWORDs
// after every 64 DWORDs (=> 272B LDS row pitch == CKP bf16 elements).
__device__ __forceinline__ void tdm_load_b_chunk(const bf16_t* gsrc, uint32_t lds_off) {
  uint64_t ga = (uint64_t)(uintptr_t)gsrc;
  u32x4 g0;
  g0[0] = 1u;                                                  // count=1 (valid user D#)
  g0[1] = lds_off;                                             // lds_addr (bytes)
  g0[2] = (uint32_t)ga;                                        // global_addr[31:0]
  g0[3] = (uint32_t)((ga >> 32) & 0x01FFFFFFu) | (2u << 30);   // global_addr[56:32], type=2
  u32x8 g1;
  g1[0] = (1u << 16)          // data_size = 2 bytes
        | (1u << 20)          // pad_enable
        | (5u << 22)          // pad_interval code 5 => every 64 DWORDs (256B)
        | (3u << 25);         // pad_amount  code 3 => 4 DWORDs (16B)
  g1[1] = ((uint32_t)(KBIG & 0xFFFFu)) << 16;                  // tensor_dim0[15:0]
  g1[2] = (uint32_t)(KBIG >> 16) | ((uint32_t)FS << 16);       // tensor_dim0[31:16] | tensor_dim1[15:0]
  g1[3] = ((uint32_t)CKB << 16);                               // tensor_dim1[31:16]=0 | tile_dim0=128
  g1[4] = (uint32_t)FS;                                        // tile_dim1=64 | tile_dim2=0
  g1[5] = (uint32_t)KBIG;                                      // tensor_dim0_stride[31:0]
  g1[6] = 0u;                                                  // stride[47:32] | dim1_stride lo
  g1[7] = 0u;
  asm volatile("tensor_load_to_lds %0, %1" :: "s"(g0), "s"(g1) : "memory");
}

__global__ __launch_bounds__(256) void k_cheb_gemm_wmma(
    const bf16_t* __restrict__ Abig,    // [N][KBIG]
    const bf16_t* __restrict__ HbigT,   // [BT][FS][KBIG]
    const float*  __restrict__ cheb_b,  // [FS]
    float* __restrict__ out_s) {        // [BT][N][FS]
  __shared__ bf16_t ldsB[2][FS][CKP];   // 2 x 64 x 136 x 2B = 34.8 KB

  const int bt    = blockIdx.x;
  const int nblk  = blockIdx.y;                 // 0..7
  const int tid   = threadIdx.x;
  const int wave  = tid >> 5;
  const int lane  = tid & 31;
  const int lhalf = lane >> 4;                  // 0 / 1
  const int l15   = lane & 15;
  const int rowbase = nblk * 128 + wave * 16;

  const bf16_t* Hbt  = HbigT + (size_t)bt * FS * KBIG;
  const bf16_t* Arow = Abig + (size_t)(rowbase + l15) * KBIG;

  const uint32_t ldsbase  = (uint32_t)(uintptr_t)(&ldsB[0][0][0]);
  const uint32_t bufbytes = (uint32_t)(FS * CKP * sizeof(bf16_t));

  v8f acc[4];
#pragma unroll
  for (int ct = 0; ct < 4; ++ct) acc[ct] = (v8f){0.f, 0.f, 0.f, 0.f, 0.f, 0.f, 0.f, 0.f};

  // preload chunk 0 into buffer 0 (one TDM issue per workgroup: wave 0 only)
  if (wave == 0) tdm_load_b_chunk(Hbt, ldsbase);

  for (int kc = 0; kc < KBIG; kc += CKB) {
    const int p = (kc >> 7) & 1;
    if (wave == 0) __builtin_amdgcn_s_wait_tensorcnt(0);
    __syncthreads();   // chunk p resident; everyone done reading buffer p^1

    // overlap: DMA next chunk into the other buffer while we compute
    if (wave == 0 && kc + CKB < KBIG)
      tdm_load_b_chunk(Hbt + kc + CKB, ldsbase + (uint32_t)(p ^ 1) * bufbytes);
    if (kc + CKB < KBIG)
      __builtin_prefetch(Arow + kc + CKB, 0, 1);

#pragma unroll
    for (int kk = 0; kk < CKB; kk += 32) {
      // A fragment: two contiguous 16B loads per lane
      union { v16bf v; float4 q[2]; } afrag;
      const float4* ap = (const float4*)(Arow + kc + kk + lhalf * 8);
      afrag.q[0] = ap[0];          // K +0..7   (or +8..15 for upper lanes)
      afrag.q[1] = ap[2];          // K +16..23 (or +24..31)

      // 4 column tiles of B from LDS
#pragma unroll
      for (int ct = 0; ct < 4; ++ct) {
        int col = ct * 16 + l15;
        union { v16bf v; float4 q[2]; } bfrag;
        const float4* bp = (const float4*)(&ldsB[p][col][kk + lhalf * 8]);
        bfrag.q[0] = bp[0];
        bfrag.q[1] = bp[2];
        acc[ct] = __builtin_amdgcn_wmma_f32_16x16x32_bf16(
            false, afrag.v, false, bfrag.v, (short)0, acc[ct], false, false);
      }
    }
  }

  // store D + cheb_b.  D layout: VGPR r, lane -> (row = r + 8*lhalf, col = l15)
  float* obase = out_s + (size_t)bt * N_ * FS;
#pragma unroll
  for (int ct = 0; ct < 4; ++ct) {
    int col = ct * 16 + l15;
    float bias = cheb_b[col];
#pragma unroll
    for (int r = 0; r < 8; ++r) {
      int row = rowbase + r + lhalf * 8;
      obase[(size_t)row * FS + col] = acc[ct][r] + bias;
    }
  }
}

// ---------------------------------------------------------------------------
// Kernel 3: temporal conv2 (causal, k=3) + GLU + LayerNorm + residual + ReLU.
// One thread per (bt, n).
// ---------------------------------------------------------------------------
__global__ __launch_bounds__(256) void k_temporal2_out(
    const float* __restrict__ out_s,    // [BT][N][FS]
    const float* __restrict__ conv2_w,  // [128][FS][3]
    const float* __restrict__ conv2_b,  // [128]
    const float* __restrict__ ln_g,     // [FO]
    const float* __restrict__ ln_b,     // [FO]
    const float* __restrict__ x,        // [BT][N][FIN]
    const float* __restrict__ res_w,    // [FO][FIN]
    const float* __restrict__ res_b,    // [FO]
    float* __restrict__ out) {          // [BT][N][FO]
  extern __shared__ float smem[];
  float* sW  = smem;                    // 128*64*3 = 24576
  float* sB  = sW + 24576;              // 128
  float* sRW = sB + 128;                // 64*32 = 2048
  float* sRB = sRW + 2048;              // 64
  float* sG  = sRB + 64;                // 64
  float* sBt = sG + 64;                 // 64
  for (int i = threadIdx.x; i < 24576; i += 256) sW[i] = conv2_w[i];
  for (int i = threadIdx.x; i < 128;   i += 256) sB[i] = conv2_b[i];
  for (int i = threadIdx.x; i < 2048;  i += 256) sRW[i] = res_w[i];
  if (threadIdx.x < 64) {
    sRB[threadIdx.x] = res_b[threadIdx.x];
    sG[threadIdx.x]  = ln_g[threadIdx.x];
    sBt[threadIdx.x] = ln_b[threadIdx.x];
  }
  __syncthreads();

  int gid = blockIdx.x * 256 + threadIdx.x;
  int bt = gid >> 10;
  int n  = gid & 1023;
  int t  = bt & (T_ - 1);

  float sv[3][FS];
#pragma unroll
  for (int j = 0; j < 3; ++j) {
    int tt = t - 2 + j;
    if (tt >= 0) {
      const float4* sp = (const float4*)(out_s + ((size_t)(bt - (2 - j)) * N_ + n) * FS);
#pragma unroll
      for (int q = 0; q < FS / 4; ++q) {
        float4 v = sp[q];
        sv[j][q * 4 + 0] = v.x; sv[j][q * 4 + 1] = v.y;
        sv[j][q * 4 + 2] = v.z; sv[j][q * 4 + 3] = v.w;
      }
    } else {
#pragma unroll
      for (int f = 0; f < FS; ++f) sv[j][f] = 0.f;
    }
  }

  float h2[FO];
  float mu = 0.f;
#pragma unroll 1
  for (int i = 0; i < FO; ++i) {
    float a = sB[i], g = sB[FO + i];
    const float* wa = sW + i * (FS * 3);
    const float* wg = sW + (FO + i) * (FS * 3);
#pragma unroll 8
    for (int f = 0; f < FS; ++f)
#pragma unroll
      for (int j = 0; j < 3; ++j) {
        float xf = sv[j][f];
        a = fmaf(wa[f * 3 + j], xf, a);
        g = fmaf(wg[f * 3 + j], xf, g);
      }
    h2[i] = a * (1.0f / (1.0f + __expf(-g)));
    mu += h2[i];
  }
  mu *= (1.0f / FO);
  float var = 0.f;
#pragma unroll
  for (int i = 0; i < FO; ++i) { float d = h2[i] - mu; var = fmaf(d, d, var); }
  float inv = __frsqrt_rn(var * (1.0f / FO) + 1e-5f);

  float xr[FIN];
  {
    const float4* xp = (const float4*)(x + ((size_t)bt * N_ + n) * FIN);
#pragma unroll
    for (int q = 0; q < FIN / 4; ++q) {
      float4 v = xp[q];
      xr[q * 4 + 0] = v.x; xr[q * 4 + 1] = v.y;
      xr[q * 4 + 2] = v.z; xr[q * 4 + 3] = v.w;
    }
  }

  float* op = out + ((size_t)bt * N_ + n) * FO;
#pragma unroll 1
  for (int o = 0; o < FO; ++o) {
    float r = sRB[o];
#pragma unroll
    for (int f = 0; f < FIN; ++f) r = fmaf(xr[f], sRW[o * FIN + f], r);
    float v = (h2[o] - mu) * inv * sG[o] + sBt[o] + r;
    op[o] = fmaxf(v, 0.f);
  }
}

// ---------------------------------------------------------------------------
// launch
// ---------------------------------------------------------------------------
extern "C" void kernel_launch(void* const* d_in, const int* in_sizes, int n_in,
                              void* d_out, int out_size, void* d_ws, size_t ws_size,
                              hipStream_t stream) {
  const float* x       = (const float*)d_in[0];
  const float* basis   = (const float*)d_in[1];
  const float* conv1_w = (const float*)d_in[2];
  const float* conv1_b = (const float*)d_in[3];
  const float* cheb_w  = (const float*)d_in[4];
  const float* cheb_b  = (const float*)d_in[5];
  const float* conv2_w = (const float*)d_in[6];
  const float* conv2_b = (const float*)d_in[7];
  const float* ln_g    = (const float*)d_in[8];
  const float* ln_b    = (const float*)d_in[9];
  const float* res_w   = (const float*)d_in[10];
  const float* res_b   = (const float*)d_in[11];
  float* out = (float*)d_out;

  char* ws = (char*)d_ws;
  bf16_t* Abig  = (bf16_t*)ws;                                     // 1024*4096*2  = 8 MB
  bf16_t* HbigT = (bf16_t*)(ws + (size_t)16 * 1024 * 1024);        // 1024*64*4096*2 = 512 MB
  float*  out_s = (float*)(ws + (size_t)(16 + 512) * 1024 * 1024); // 1024*1024*64*4 = 256 MB

  // K0: basis -> Abig (bf16, row-major)
  k_convert_basis<<<dim3((KORD * N_ * N_) / 256), dim3(256), 0, stream>>>(basis, Abig);

  // K1: conv1+GLU + cheb_w fold -> HbigT (bf16, column-major over K)
  size_t shmem1 = (size_t)(12288 + 128 + 16384) * sizeof(float);   // 115200 B
  k_temporal1_fold<<<dim3((BT * N_) / 256), dim3(256), shmem1, stream>>>(
      x, conv1_w, conv1_b, cheb_w, HbigT);

  // K2: WMMA GEMM (TDM-staged B) -> out_s
  k_cheb_gemm_wmma<<<dim3(BT, N_ / 128), dim3(256), 0, stream>>>(
      Abig, HbigT, cheb_b, out_s);

  // K3: conv2+GLU + LayerNorm + residual + ReLU -> out
  size_t shmem3 = (size_t)(24576 + 128 + 2048 + 64 + 64 + 64) * sizeof(float); // ~108 KB
  k_temporal2_out<<<dim3((BT * N_) / 256), dim3(256), shmem3, stream>>>(
      out_s, conv2_w, conv2_b, ln_g, ln_b, x, res_w, res_b, out);
}
